// gcn1_17454747090955
// MI455X (gfx1250) — matrix operand order repaired and checked by reading, not compiled, not run on hardware
//
#include <hip/hip_runtime.h>

// GCN block, all matrix work on WMMA (bf16, fp32 accum):
//   out = x + LN( mix(A, mix(A, LN(x)) @ W1cat) @ W2cat )
// Adjacency mixes run on the matrix pipe via operand swap; GEMM tiles are fed
// by the Tensor Data Mover (tensor_load_to_lds, double-buffered, TENSORcnt).

typedef __attribute__((ext_vector_type(16))) __bf16 v16bf;
typedef __attribute__((ext_vector_type(8)))  float  v8f;
typedef __attribute__((ext_vector_type(4)))  unsigned int v4u;
typedef __attribute__((ext_vector_type(8)))  int v8i;
typedef __attribute__((ext_vector_type(4)))  int v4i;

#define BATCH 8192
#define NV 17
#define NC 256
#define NH 512
#define NK 3
#define MTOT (BATCH*NV)   // 139264
#define KD1 (NK*NC)       // 768
#define KD2 (NK*NH)       // 1536
#define EPSV 1e-6f

__device__ __forceinline__ float wave_sum(float v) {
#pragma unroll
  for (int off = 16; off > 0; off >>= 1) v += __shfl_xor(v, off, 32);
  return v;
}

union Frag { v16bf v; uint4 q[2]; };
union Pack8 { __bf16 e[8]; uint4 q; };

// TDM: DMA a (tile_rows x 32 bf16) tile from global (row stride stride_elems) into
// LDS at lds_off, padding each 64B row by 16B -> 80B pitch (40 bf16, matches frags).
__device__ __forceinline__ void tdm_load_tile(unsigned lds_off, const __bf16* gptr,
                                              unsigned tile_rows, unsigned stride_elems) {
  unsigned long long ga = (unsigned long long)(size_t)gptr;
  v4u g0;
  g0[0] = 1u;                                              // count=1, user mode
  g0[1] = lds_off;                                         // LDS byte address
  g0[2] = (unsigned)(ga & 0xFFFFFFFFu);                    // global_addr[31:0]
  g0[3] = (unsigned)((ga >> 32) & 0x01FFFFFFu) | (2u << 30); // addr[56:32] | type=2
  v8i g1;
  g1[0] = (int)((1u << 16) | (1u << 20) | (3u << 22) | (3u << 25));
  //            data_size=2B  pad_en      pad_int=16dw  pad_amt=4dw
  g1[1] = (int)((32u & 0xFFFFu) << 16);                    // tensor_dim0 lo16 (=32)
  g1[2] = (int)((32u >> 16) | ((tile_rows & 0xFFFFu) << 16)); // dim0 hi | tensor_dim1 lo
  g1[3] = (int)((tile_rows >> 16) | (32u << 16));          // tensor_dim1 hi | tile_dim0=32
  g1[4] = (int)(tile_rows & 0xFFFFu);                      // tile_dim1 | tile_dim2=0
  g1[5] = (int)stride_elems;                               // tensor_dim0_stride lo32
  g1[6] = 0;                                               // stride0 hi | stride1 lo
  g1[7] = 0;
  v4i z4 = {0, 0, 0, 0};
#if defined(__clang_major__) && (__clang_major__ >= 23)
  v8i z8 = {0, 0, 0, 0, 0, 0, 0, 0};
  __builtin_amdgcn_tensor_load_to_lds(g0, g1, z4, z4, z8, 0);
#else
  __builtin_amdgcn_tensor_load_to_lds(g0, g1, z4, z4, 0);
#endif
}

__device__ __forceinline__ unsigned lds_off_u32(const void* p) {
  return (unsigned)(size_t)p;    // generic LDS address: low 32 bits = LDS byte offset
}

// ---------------------------------------------------------------- weight prep
__global__ __launch_bounds__(256) void prep_w1_kernel(const float* __restrict__ W1,
                                                      __bf16* __restrict__ Wc1) {
  int idx = blockIdx.x * 256 + threadIdx.x;
  if (idx >= NH * KD1) return;
  int h = idx / KD1, kc = idx % KD1, k = kc >> 8, c = kc & 255;
  Wc1[idx] = (__bf16)W1[(size_t)(k * NH + h) * NC + c];
}

__global__ __launch_bounds__(256) void prep_w2_kernel(const float* __restrict__ W2,
                                                      __bf16* __restrict__ Wc2) {
  int idx = blockIdx.x * 256 + threadIdx.x;
  if (idx >= NC * KD2) return;
  int c = idx / KD2, kh = idx % KD2, k = kh >> 9, h = kh & 511;
  Wc2[idx] = (__bf16)W2[(size_t)(k * NC + c) * NH + h];
}

// Abt[k][w][v] = A[k,v,w] zero-padded to 32x32; bias folded through sum_v A[k,v,w].
__global__ __launch_bounds__(256) void prep_misc_kernel(const float* __restrict__ A,
                                                        const float* __restrict__ c1,
                                                        const float* __restrict__ c2,
                                                        __bf16* __restrict__ Abt,
                                                        float* __restrict__ bias1,
                                                        float* __restrict__ bias2) {
  __shared__ float Asum[NK * NV];
  if (threadIdx.x < NK * NV) {
    int k = threadIdx.x / NV, w = threadIdx.x % NV;
    float s = 0.f;
    for (int v = 0; v < NV; ++v) s += A[k * NV * NV + v * NV + w];
    Asum[threadIdx.x] = s;
  }
  __syncthreads();
  for (int i = threadIdx.x; i < NK * 32 * 32; i += 256) {
    int k = i >> 10, w = (i >> 5) & 31, v = i & 31;
    float val = (v < NV && w < NV) ? A[k * NV * NV + v * NV + w] : 0.f;
    Abt[i] = (__bf16)val;
  }
  for (int i = threadIdx.x; i < NV * NH; i += 256) {
    int w = i / NH, h = i % NH;
    float s = 0.f;
    for (int k = 0; k < NK; ++k) s += c1[k * NH + h] * Asum[k * NV + w];
    bias1[i] = s;
  }
  for (int i = threadIdx.x; i < NV * NC; i += 256) {
    int w = i / NC, c = i % NC;
    float s = 0.f;
    for (int k = 0; k < NK; ++k) s += c2[k * NC + c] * Asum[k * NV + w];
    bias2[i] = s;
  }
}

// ------------------------------------------------- LN(x) + WMMA mix1 -> Tcat1
__global__ __launch_bounds__(256) void ln_mix1_kernel(const float* __restrict__ x,
                                                      const float* __restrict__ g,
                                                      const float* __restrict__ b,
                                                      const __bf16* __restrict__ Abt,
                                                      __bf16* __restrict__ Tcat1) {
  __shared__ __align__(16) float xs[NV * NC];
  __shared__ __align__(16) __bf16 tn[NC * 32];        // [c][vpad]
  __shared__ __align__(16) __bf16 Ab_s[NK * 32 * 32]; // [k][w][v]
  const int tid = threadIdx.x;
  const int bs = blockIdx.x;

  const float4* xp = (const float4*)(x + (size_t)bs * NV * NC);
  for (int i = tid; i < NV * NC / 4; i += 256) ((float4*)xs)[i] = xp[i];
  for (int i = tid; i < NK * 32 * 32 / 8; i += 256)
    ((uint4*)Ab_s)[i] = ((const uint4*)Abt)[i];
  for (int i = tid; i < NC * 15; i += 256) {          // zero pad columns v=17..31
    int c = i / 15, v = NV + i % 15;
    tn[c * 32 + v] = (__bf16)0.f;
  }
  __syncthreads();

  const int wid = tid >> 5, lane = tid & 31;
  for (int v = wid; v < NV; v += 8) {
    const float* xr = &xs[v * NC + lane * 8];
    float s = 0.f;
#pragma unroll
    for (int j = 0; j < 8; ++j) s += xr[j];
    s = wave_sum(s);
    float mu = s * (1.f / NC);
    float q = 0.f;
#pragma unroll
    for (int j = 0; j < 8; ++j) { float d = xr[j] - mu; q += d * d; }
    q = wave_sum(q);
    float rinv = rsqrtf(q * (1.f / NC) + EPSV);
#pragma unroll
    for (int j = 0; j < 8; ++j) {
      int c = lane * 8 + j;
      tn[c * 32 + v] = (__bf16)((xr[j] - mu) * rinv * g[c] + b[c]);
    }
  }
  __syncthreads();

  const int ml = lane & 15, hh = lane >> 4;
  const v8f vzero = {0.f, 0.f, 0.f, 0.f, 0.f, 0.f, 0.f, 0.f};
  Frag fb[NK][2];
#pragma unroll
  for (int k = 0; k < NK; ++k)
#pragma unroll
    for (int nt = 0; nt < 2; ++nt) {
      fb[k][nt].q[0] = *(const uint4*)&Ab_s[k * 1024 + (nt * 16 + ml) * 32 + hh * 8];
      fb[k][nt].q[1] = *(const uint4*)&Ab_s[k * 1024 + (nt * 16 + ml) * 32 + 16 + hh * 8];
    }
#pragma unroll
  for (int t = 0; t < 2; ++t) {
    int mt = wid * 2 + t;                             // c-tile
    Frag fa;
    fa.q[0] = *(const uint4*)&tn[(mt * 16 + ml) * 32 + hh * 8];
    fa.q[1] = *(const uint4*)&tn[(mt * 16 + ml) * 32 + 16 + hh * 8];
#pragma unroll
    for (int k = 0; k < NK; ++k)
#pragma unroll
      for (int nt = 0; nt < 2; ++nt) {
        v8f acc = __builtin_amdgcn_wmma_f32_16x16x32_bf16(
            false, fa.v, false, fb[k][nt].v, (short)0, vzero, false, false);
        int w = nt * 16 + ml;
        if (w < NV) {                                 // acc rows = 8 consecutive c
          Pack8 o;
#pragma unroll
          for (int r = 0; r < 8; ++r) o.e[r] = (__bf16)acc[r];
          *(uint4*)&Tcat1[(size_t)(bs * NV + w) * KD1 + k * NC + mt * 16 + hh * 8] = o.q;
        }
      }
  }
}

// ------------------------------------------------------------- GEMM1 (TDM-fed)
// y1[M,512] = Tcat1[M,768] @ Wc1^T + bias1 ; 128x128 block, 8 waves of 32x64.
#define G1_MT 128
#define G1_NT 128
__global__ __launch_bounds__(256) void gemm1_kernel(const __bf16* __restrict__ Tcat1,
                                                    const __bf16* __restrict__ Wc1,
                                                    const float* __restrict__ bias1,
                                                    __bf16* __restrict__ y1) {
  __shared__ __align__(16) __bf16 As[2 * G1_MT * 40];
  __shared__ __align__(16) __bf16 Bs[2 * G1_NT * 40];
  const int tid = threadIdx.x;
  const int m0 = blockIdx.y * G1_MT;
  const int n0 = blockIdx.x * G1_NT;
  const int wid = tid >> 5, lane = tid & 31;
  const int wm = wid & 3, wn = wid >> 2;
  const int ml = lane & 15, hh = lane >> 4;
  const unsigned asOff = lds_off_u32(As), bsOff = lds_off_u32(Bs);
  const int STEPS = KD1 / 32;

  v8f acc[2][4];
  const v8f vzero = {0.f, 0.f, 0.f, 0.f, 0.f, 0.f, 0.f, 0.f};
#pragma unroll
  for (int i = 0; i < 2; ++i)
#pragma unroll
    for (int j = 0; j < 4; ++j) acc[i][j] = vzero;

  if (wid == 0) {                                     // prime buffer 0
    tdm_load_tile(asOff, Tcat1 + (size_t)m0 * KD1, G1_MT, KD1);
    tdm_load_tile(bsOff, Wc1 + (size_t)n0 * KD1, G1_NT, KD1);
  }

  for (int kt = 0; kt < STEPS; ++kt) {
    __syncthreads();                                  // all done reading buf[kt-1]
    if (wid == 0) {
      if (kt + 1 < STEPS) {                           // prefetch buf[kt+1] via TDM
        unsigned bn = (unsigned)((kt + 1) & 1);
        tdm_load_tile(asOff + bn * G1_MT * 80, Tcat1 + (size_t)m0 * KD1 + (kt + 1) * 32,
                      G1_MT, KD1);
        tdm_load_tile(bsOff + bn * G1_NT * 80, Wc1 + (size_t)n0 * KD1 + (kt + 1) * 32,
                      G1_NT, KD1);
        __builtin_amdgcn_s_wait_tensorcnt(2);         // buf[kt] complete
      } else {
        __builtin_amdgcn_s_wait_tensorcnt(0);
      }
    }
    __syncthreads();                                  // buf[kt] visible to all waves

    const __bf16* Ab = As + (kt & 1) * G1_MT * 40;
    const __bf16* Bb = Bs + (kt & 1) * G1_NT * 40;
    Frag fa[2], fb[4];
#pragma unroll
    for (int sm = 0; sm < 2; ++sm) {
      int r = wm * 32 + sm * 16 + ml;
      fa[sm].q[0] = *(const uint4*)&Ab[r * 40 + hh * 8];
      fa[sm].q[1] = *(const uint4*)&Ab[r * 40 + 16 + hh * 8];
    }
#pragma unroll
    for (int sn = 0; sn < 4; ++sn) {
      int n = wn * 64 + sn * 16 + ml;
      fb[sn].q[0] = *(const uint4*)&Bb[n * 40 + hh * 8];
      fb[sn].q[1] = *(const uint4*)&Bb[n * 40 + 16 + hh * 8];
    }
#pragma unroll
    for (int sm = 0; sm < 2; ++sm)
#pragma unroll
      for (int sn = 0; sn < 4; ++sn)
        acc[sm][sn] = __builtin_amdgcn_wmma_f32_16x16x32_bf16(
            false, fa[sm].v, false, fb[sn].v, (short)0, acc[sm][sn], false, false);
  }

#pragma unroll
  for (int sm = 0; sm < 2; ++sm)
#pragma unroll
    for (int sn = 0; sn < 4; ++sn) {
      int n = n0 + wn * 64 + sn * 16 + ml;
#pragma unroll
      for (int r = 0; r < 8; ++r) {
        int m = m0 + wm * 32 + sm * 16 + hh * 8 + r;
        float v = acc[sm][sn][r] + bias1[(m % NV) * NH + n];
        y1[(size_t)m * NH + n] = (__bf16)v;
      }
    }
}

// ------------------------------------------------- WMMA mix2: y1 -> Tcat2
__global__ __launch_bounds__(256) void mix2_kernel(const __bf16* __restrict__ y1,
                                                   const __bf16* __restrict__ Abt,
                                                   __bf16* __restrict__ Tcat2) {
  __shared__ __align__(16) __bf16 yt[NH * 32];        // [h][vpad]
  __shared__ __align__(16) __bf16 Ab_s[NK * 32 * 32];
  const int tid = threadIdx.x;
  const int bs = blockIdx.x;

  for (int i = tid; i < NK * 32 * 32 / 8; i += 256)
    ((uint4*)Ab_s)[i] = ((const uint4*)Abt)[i];
  for (int i = tid; i < NH * 15; i += 256) {          // zero pad v=17..31
    int h = i / 15, v = NV + i % 15;
    yt[h * 32 + v] = (__bf16)0.f;
  }
  for (int i = tid; i < NV * NH / 8; i += 256) {      // load + transpose into LDS
    int v = i >> 6, h8 = (i & 63) * 8;
    Pack8 u;
    u.q = *(const uint4*)&y1[((size_t)bs * NV + v) * NH + h8];
#pragma unroll
    for (int j = 0; j < 8; ++j) yt[(h8 + j) * 32 + v] = u.e[j];
  }
  __syncthreads();

  const int wid = tid >> 5, lane = tid & 31;
  const int ml = lane & 15, hh = lane >> 4;
  const v8f vzero = {0.f, 0.f, 0.f, 0.f, 0.f, 0.f, 0.f, 0.f};
  Frag fb[NK][2];
#pragma unroll
  for (int k = 0; k < NK; ++k)
#pragma unroll
    for (int nt = 0; nt < 2; ++nt) {
      fb[k][nt].q[0] = *(const uint4*)&Ab_s[k * 1024 + (nt * 16 + ml) * 32 + hh * 8];
      fb[k][nt].q[1] = *(const uint4*)&Ab_s[k * 1024 + (nt * 16 + ml) * 32 + 16 + hh * 8];
    }
#pragma unroll
  for (int t = 0; t < 4; ++t) {
    int mt = wid * 4 + t;                             // h-tile
    Frag fa;
    fa.q[0] = *(const uint4*)&yt[(mt * 16 + ml) * 32 + hh * 8];
    fa.q[1] = *(const uint4*)&yt[(mt * 16 + ml) * 32 + 16 + hh * 8];
#pragma unroll
    for (int k = 0; k < NK; ++k)
#pragma unroll
      for (int nt = 0; nt < 2; ++nt) {
        v8f acc = __builtin_amdgcn_wmma_f32_16x16x32_bf16(
            false, fa.v, false, fb[k][nt].v, (short)0, vzero, false, false);
        int w = nt * 16 + ml;
        if (w < NV) {
          Pack8 o;
#pragma unroll
          for (int r = 0; r < 8; ++r) o.e[r] = (__bf16)acc[r];
          *(uint4*)&Tcat2[(size_t)(bs * NV + w) * KD2 + k * NH + mt * 16 + hh * 8] = o.q;
        }
      }
  }
}

// --------------------------- GEMM2 (TDM-fed) + bias + LayerNorm + residual
#define G2_MT 64
#define G2_NT 256
__global__ __launch_bounds__(256) void gemm2_kernel(const __bf16* __restrict__ Tcat2,
                                                    const __bf16* __restrict__ Wc2,
                                                    const float* __restrict__ bias2,
                                                    const float* __restrict__ x,
                                                    const float* __restrict__ g2,
                                                    const float* __restrict__ b2,
                                                    float* __restrict__ out) {
  __shared__ __align__(16) unsigned char raw[G2_MT * 264 * 4];  // 67584 B
  __bf16* As = (__bf16*)raw;                   // 2 x 64*40*2  = 10240
  __bf16* Bs = (__bf16*)(raw + 10240);         // 2 x 256*40*2 = 40960 (total 51200)
  float* Ct = (float*)raw;                     // [64][264] after the k-loop

  const int tid = threadIdx.x;
  const int m0 = blockIdx.x * G2_MT;
  const int wid = tid >> 5, lane = tid & 31;
  const int wm = wid & 1, wn = wid >> 1;
  const int ml = lane & 15, hh = lane >> 4;
  const unsigned asOff = lds_off_u32(As), bsOff = lds_off_u32(Bs);
  const int STEPS = KD2 / 32;

  v8f acc[2][4];
  const v8f vzero = {0.f, 0.f, 0.f, 0.f, 0.f, 0.f, 0.f, 0.f};
#pragma unroll
  for (int i = 0; i < 2; ++i)
#pragma unroll
    for (int j = 0; j < 4; ++j) acc[i][j] = vzero;

  if (wid == 0) {
    tdm_load_tile(asOff, Tcat2 + (size_t)m0 * KD2, G2_MT, KD2);
    tdm_load_tile(bsOff, Wc2, G2_NT, KD2);
  }

  for (int kt = 0; kt < STEPS; ++kt) {
    __syncthreads();
    if (wid == 0) {
      if (kt + 1 < STEPS) {
        unsigned bn = (unsigned)((kt + 1) & 1);
        tdm_load_tile(asOff + bn * G2_MT * 80, Tcat2 + (size_t)m0 * KD2 + (kt + 1) * 32,
                      G2_MT, KD2);
        tdm_load_tile(bsOff + bn * G2_NT * 80, Wc2 + (size_t)(kt + 1) * 32, G2_NT, KD2);
        __builtin_amdgcn_s_wait_tensorcnt(2);
      } else {
        __builtin_amdgcn_s_wait_tensorcnt(0);
      }
    }
    __syncthreads();

    const __bf16* Ab = As + (kt & 1) * G2_MT * 40;
    const __bf16* Bb = Bs + (kt & 1) * G2_NT * 40;
    Frag fa[2], fb[4];
#pragma unroll
    for (int sm = 0; sm < 2; ++sm) {
      int r = wm * 32 + sm * 16 + ml;
      fa[sm].q[0] = *(const uint4*)&Ab[r * 40 + hh * 8];
      fa[sm].q[1] = *(const uint4*)&Ab[r * 40 + 16 + hh * 8];
    }
#pragma unroll
    for (int sn = 0; sn < 4; ++sn) {
      int n = wn * 64 + sn * 16 + ml;
      fb[sn].q[0] = *(const uint4*)&Bb[n * 40 + hh * 8];
      fb[sn].q[1] = *(const uint4*)&Bb[n * 40 + 16 + hh * 8];
    }
#pragma unroll
    for (int sm = 0; sm < 2; ++sm)
#pragma unroll
      for (int sn = 0; sn < 4; ++sn)
        acc[sm][sn] = __builtin_amdgcn_wmma_f32_16x16x32_bf16(
            false, fa[sm].v, false, fb[sn].v, (short)0, acc[sm][sn], false, false);
  }

  __syncthreads();                               // retire all As/Bs readers, reuse raw as Ct
#pragma unroll
  for (int sm = 0; sm < 2; ++sm)
#pragma unroll
    for (int sn = 0; sn < 4; ++sn) {
      int n = wn * 64 + sn * 16 + ml;
#pragma unroll
      for (int r = 0; r < 8; ++r) {
        int row = wm * 32 + sm * 16 + hh * 8 + r;
        Ct[row * 264 + n] = acc[sm][sn][r] + bias2[((m0 + row) % NV) * NC + n];
      }
    }
  __syncthreads();

  for (int row = wid; row < G2_MT; row += 8) {
    float s = 0.f;
    for (int c = lane; c < NC; c += 32) s += Ct[row * 264 + c];
    s = wave_sum(s);
    float mu = s * (1.f / NC);
    float q = 0.f;
    for (int c = lane; c < NC; c += 32) { float d = Ct[row * 264 + c] - mu; q += d * d; }
    q = wave_sum(q);
    float rinv = rsqrtf(q * (1.f / NC) + EPSV);
    size_t m = (size_t)(m0 + row);
    for (int c = lane; c < NC; c += 32)
      out[m * NC + c] = x[m * NC + c] + (Ct[row * 264 + c] - mu) * rinv * g2[c] + b2[c];
  }
}

// ---------------------------------------------------------------- launch
extern "C" void kernel_launch(void* const* d_in, const int* in_sizes, int n_in,
                              void* d_out, int out_size, void* d_ws, size_t ws_size,
                              hipStream_t stream) {
  (void)in_sizes; (void)n_in; (void)out_size; (void)ws_size;
  const float* x  = (const float*)d_in[0];
  const float* A  = (const float*)d_in[1];
  const float* g1 = (const float*)d_in[2];
  const float* b1 = (const float*)d_in[3];
  const float* W1 = (const float*)d_in[4];
  const float* c1 = (const float*)d_in[5];
  const float* W2 = (const float*)d_in[6];
  const float* c2 = (const float*)d_in[7];
  const float* g2 = (const float*)d_in[8];
  const float* b2 = (const float*)d_in[9];
  float* out = (float*)d_out;

  char* ws = (char*)d_ws;
  // Tcat1 (live: ln_mix1..gemm1) overlaps Tcat2 (live: mix2..gemm2).
  __bf16* Tcat2 = (__bf16*)ws;                                   // 427.8 MB
  __bf16* Tcat1 = (__bf16*)ws;                                   // 213.9 MB (same region)
  size_t off = (size_t)MTOT * KD2 * 2;
  __bf16* y1   = (__bf16*)(ws + off); off += (size_t)MTOT * NH * 2;  // 142.6 MB
  __bf16* Wc1  = (__bf16*)(ws + off); off += (size_t)NH * KD1 * 2;
  __bf16* Wc2  = (__bf16*)(ws + off); off += (size_t)NC * KD2 * 2;
  __bf16* Abt  = (__bf16*)(ws + off); off += (size_t)NK * 32 * 32 * 2;
  float* bias1 = (float*)(ws + off);  off += (size_t)NV * NH * 4;
  float* bias2 = (float*)(ws + off);  off += (size_t)NV * NC * 4;

  prep_w1_kernel<<<(NH * KD1 + 255) / 256, 256, 0, stream>>>(W1, Wc1);
  prep_w2_kernel<<<(NC * KD2 + 255) / 256, 256, 0, stream>>>(W2, Wc2);
  prep_misc_kernel<<<1, 256, 0, stream>>>(A, c1, c2, Abt, bias1, bias2);
  ln_mix1_kernel<<<BATCH, 256, 0, stream>>>(x, g1, b1, Abt, Tcat1);
  gemm1_kernel<<<dim3(NH / G1_NT, MTOT / G1_MT), 256, 0, stream>>>(Tcat1, Wc1, bias1, y1);
  mix2_kernel<<<BATCH, 256, 0, stream>>>(y1, Abt, Tcat2);
  gemm2_kernel<<<MTOT / G2_MT, 256, 0, stream>>>(Tcat2, Wc2, bias2, x, g2, b2, out);
}
